// CondConv3d_59854664237937
// MI455X (gfx1250) — matrix operand order repaired and checked by reading, not compiled, not run on hardware
//
#include <hip/hip_runtime.h>

typedef _Float16 v16h __attribute__((ext_vector_type(16)));
typedef float    v8f  __attribute__((ext_vector_type(8)));

namespace {
constexpr int kB = 8, kCI = 32, kCO = 32, kD = 16, kH = 64, kW = 64;
constexpr int kE = 8, kTap = 27;
constexpr int TH = 4, TW = 32;                 // output tile per workgroup: 4h x 32w
constexpr int PIXH = TH + 2, PIXW = TW + 2;    // halo
constexpr int PIX = 3 * PIXH * PIXW;           // 612 staged input "pixels"
constexpr int DHW = kD * kH * kW;              // 65536
constexpr int AELEMS = kTap * 2 * 32 * 16;     // 27648 pre-swizzled A halves
constexpr size_t WS_AHI  = 0;
constexpr size_t WS_ALO  = (size_t)AELEMS * 2;       // 55296 B
constexpr size_t WS_BSUM = (size_t)AELEMS * 4;       // 110592 B
constexpr size_t WS_SSC  = WS_BSUM + 32 * 4;
}

// ---------------------------------------------------------------------------
// Prep 1: Wsum = sum_e weight[e], hi/lo f16 split, pre-swizzled into the exact
// per-lane VGPR order of the WMMA A operand: idx = ((tap*2+mtile)*32+lane)*16+j
// ---------------------------------------------------------------------------
__global__ void prep_weights(const float* __restrict__ weight,
                             _Float16* __restrict__ a_hi,
                             _Float16* __restrict__ a_lo) {
  int idx = blockIdx.x * 256 + threadIdx.x;
  if (idx >= AELEMS) return;
  int j     = idx & 15;
  int lane  = (idx >> 4) & 31;
  int mtile = (idx >> 9) & 1;
  int tap   = idx >> 10;
  int m = mtile * 16 + (lane & 15);
  // A layout (matches B staging): lane<16: K = j ; lane>=16: K = j+16
  int c = (lane < 16) ? j : j + 16;
  float s = 0.f;
  #pragma unroll
  for (int e = 0; e < kE; ++e)
    s += weight[(((size_t)e * kCO + m) * kCI + c) * kTap + tap];
  _Float16 hi = (_Float16)s;
  a_hi[idx] = hi;
  a_lo[idx] = (_Float16)(s - (float)hi);
}

// ---------------------------------------------------------------------------
// Prep 2: bsum[o] = sum_e bias[e,o]; ssc[b] = sum_e routing_weights[b,e]
// ---------------------------------------------------------------------------
__global__ void prep_bias(const float* __restrict__ rw,
                          const float* __restrict__ bias,
                          float* __restrict__ bsum,
                          float* __restrict__ ssc) {
  int t = threadIdx.x;
  if (t < kCO) {
    float s = 0.f;
    #pragma unroll
    for (int e = 0; e < kE; ++e) s += bias[e * kCO + t];
    bsum[t] = s;
  } else if (t < kCO + kB) {
    int b = t - kCO;
    float s = 0.f;
    #pragma unroll
    for (int e = 0; e < kE; ++e) s += rw[b * kE + e];
    ssc[b] = s;
  }
}

// ---------------------------------------------------------------------------
// Main: implicit-GEMM 3D conv on v_wmma_f32_16x16x32_f16 with hi/lo f16
// compensation.  M=C_OUT(2 tiles of 16), N=spatial (8 tiles of 16 per WG),
// K = 27 taps x 32 channels (one WMMA K-slab per tap).
// ---------------------------------------------------------------------------
__launch_bounds__(256)
__global__ void condconv_main(const float* __restrict__ x,
                              const _Float16* __restrict__ a_hi_g,
                              const _Float16* __restrict__ a_lo_g,
                              const float* __restrict__ bsum,
                              const float* __restrict__ ssc,
                              float* __restrict__ out) {
  __shared__ _Float16 sXhi[PIX * kCI];   // 39168 B
  __shared__ _Float16 sXlo[PIX * kCI];   // 39168 B

  const int tid = threadIdx.x;
  const int bid = blockIdx.x;
  const int wt = bid & 1;
  const int ht = (bid >> 1) & 15;
  const int d  = (bid >> 5) & 15;
  const int b  = bid >> 9;
  const int h0 = ht * TH;
  const int w0 = wt * TW;

  // ---- stage input halo region to LDS, layout [pixel][c], hi/lo split ----
  for (int p = tid; p < PIX; p += 256) {
    int wr   = p % PIXW;
    int rest = p / PIXW;
    int hz   = rest % PIXH;
    int dz   = rest / PIXH;
    int dp = d + dz - 1;
    int hp = h0 + hz - 1;
    int wp = w0 + wr - 1;
    bool ok = (dp >= 0) & (dp < kD) & (hp >= 0) & (hp < kH) & (wp >= 0) & (wp < kW);
    const float* xb = x + (size_t)b * kCI * DHW + (size_t)dp * (kH * kW) + hp * kW + wp;
    #pragma unroll
    for (int c = 0; c < kCI; ++c) {
      float v = ok ? xb[(size_t)c * DHW] : 0.f;
      _Float16 hi = (_Float16)v;
      sXhi[p * kCI + c] = hi;
      sXlo[p * kCI + c] = (_Float16)(v - (float)hi);
    }
  }
  __syncthreads();

  const int lane  = tid & 31;
  const int wv    = tid >> 5;
  const int mtile = wv & 1;        // same M-tile for both of this wave's tiles
  const int nt0   = wv >> 1;       // N-tiles nt0 and nt0+4
  const int nl    = lane & 15;
  const int boff  = (lane < 16) ? 0 : 16;   // B operand: lanes>=16 hold K=16..31

  v8f acc0 = {};
  v8f acc1 = {};

  const v16h* __restrict__ Ah = (const v16h*)a_hi_g;
  const v16h* __restrict__ Al = (const v16h*)a_lo_g;

  for (int tap = 0; tap < kTap; ++tap) {
    const int kw = tap % 3;
    const int kh = (tap / 3) % 3;
    const int kd = tap / 9;

    // A tile (shared by both N-tiles): one contiguous 32B load per lane, L2-resident
    v16h ahi = Ah[(tap * 2 + mtile) * 32 + lane];
    v16h alo = Al[(tap * 2 + mtile) * 32 + lane];

    // ---- N-tile 0 ----
    {
      const int h_off = nt0 >> 1;
      const int w_in  = (nt0 & 1) * 16 + nl;
      const int q = (kd * PIXH + h_off + kh) * PIXW + (w_in + kw);
      v16h bhi = *(const v16h*)&sXhi[q * kCI + boff];
      v16h blo = *(const v16h*)&sXlo[q * kCI + boff];
      acc0 = __builtin_amdgcn_wmma_f32_16x16x32_f16(false, ahi, false, bhi, (short)0, acc0, false, false);
      acc0 = __builtin_amdgcn_wmma_f32_16x16x32_f16(false, ahi, false, blo, (short)0, acc0, false, false);
      acc0 = __builtin_amdgcn_wmma_f32_16x16x32_f16(false, alo, false, bhi, (short)0, acc0, false, false);
    }
    // ---- N-tile 1 ----
    {
      const int nt = nt0 + 4;
      const int h_off = nt >> 1;
      const int w_in  = (nt & 1) * 16 + nl;
      const int q = (kd * PIXH + h_off + kh) * PIXW + (w_in + kw);
      v16h bhi = *(const v16h*)&sXhi[q * kCI + boff];
      v16h blo = *(const v16h*)&sXlo[q * kCI + boff];
      acc1 = __builtin_amdgcn_wmma_f32_16x16x32_f16(false, ahi, false, bhi, (short)0, acc1, false, false);
      acc1 = __builtin_amdgcn_wmma_f32_16x16x32_f16(false, ahi, false, blo, (short)0, acc1, false, false);
      acc1 = __builtin_amdgcn_wmma_f32_16x16x32_f16(false, alo, false, bhi, (short)0, acc1, false, false);
    }
  }

  // ---- epilogue: out = s_b * (acc + bsum[o]) ----
  const float sb = ssc[b];
  const int mrow = (lane < 16) ? 0 : 8;   // C/D layout: VGPR r -> M = r (+8 for hi lanes)

  {
    const int h = h0 + (nt0 >> 1);
    const int w = w0 + (nt0 & 1) * 16 + nl;
    #pragma unroll
    for (int r = 0; r < 8; ++r) {
      int o = mtile * 16 + mrow + r;
      out[((((size_t)b * kCO + o) * kD + d) * kH + h) * kW + w] = sb * (acc0[r] + bsum[o]);
    }
  }
  {
    const int nt = nt0 + 4;
    const int h = h0 + (nt >> 1);
    const int w = w0 + (nt & 1) * 16 + nl;
    #pragma unroll
    for (int r = 0; r < 8; ++r) {
      int o = mtile * 16 + mrow + r;
      out[((((size_t)b * kCO + o) * kD + d) * kH + h) * kW + w] = sb * (acc1[r] + bsum[o]);
    }
  }
}

extern "C" void kernel_launch(void* const* d_in, const int* in_sizes, int n_in,
                              void* d_out, int out_size, void* d_ws, size_t ws_size,
                              hipStream_t stream) {
  (void)in_sizes; (void)n_in; (void)out_size; (void)ws_size;
  const float* x      = (const float*)d_in[0];
  const float* rw     = (const float*)d_in[1];
  const float* weight = (const float*)d_in[2];
  const float* bias   = (const float*)d_in[3];

  char* ws = (char*)d_ws;
  _Float16* a_hi = (_Float16*)(ws + WS_AHI);
  _Float16* a_lo = (_Float16*)(ws + WS_ALO);
  float* bsum    = (float*)(ws + WS_BSUM);
  float* ssc     = (float*)(ws + WS_SSC);
  float* out     = (float*)d_out;

  prep_weights<<<(AELEMS + 255) / 256, 256, 0, stream>>>(weight, a_hi, a_lo);
  prep_bias<<<1, 64, 0, stream>>>(rw, bias, bsum, ssc);

  const int nblk = kB * kD * (kH / TH) * (kW / TW);   // 4096 workgroups
  condconv_main<<<nblk, 256, 0, stream>>>(x, a_hi, a_lo, bsum, ssc, out);
}
